// Attention_21775484191484
// MI455X (gfx1250) — compile-verified, matrix operand order checked by reading, not run on hardware
//
#include <hip/hip_runtime.h>
#include <hip/hip_bf16.h>
#include <math.h>

// Problem constants
#define BB   2
#define NQ   1024
#define NKV  2048
#define DM   1024
#define NH   16
#define DH   64
#define HD   (NH * DH)       // 1024
#define SCL  0.125f          // DH^-0.5

typedef __bf16 bf16;
typedef __attribute__((ext_vector_type(16))) __bf16 v16bf;
typedef __attribute__((ext_vector_type(8)))  float  v8f;

// ---------------------------------------------------------------------------
// WMMA helpers (CDNA5 v_wmma_f32_16x16x32_bf16, wave32)
// 16-bit A-matrix 16x32 VGPR layout (ISA 7.12.2):
//   lane<16 : row=lane,    halves K[0..7]   then K[16..23]
//   lane>=16: row=lane-16, halves K[8..15]  then K[24..31]
// B-matrix 32x16 is symmetric with N<->M, so the same gather works on a
// [N][K] (row-major, n-major) LDS tile.
// ---------------------------------------------------------------------------
__device__ __forceinline__ v16bf load_frag(const bf16* base, int ld) {
  const int lane = threadIdx.x & 31;
  const int r    = lane & 15;
  const int k0   = (lane & 16) ? 8 : 0;
  const bf16* p  = base + r * ld + k0;
  v16bf f;
#pragma unroll
  for (int i = 0; i < 8; ++i) f[i] = p[i];
#pragma unroll
  for (int i = 0; i < 8; ++i) f[8 + i] = p[16 + i];
  return f;
}

__device__ __forceinline__ v8f wmma_bf16(v16bf a, v16bf b, v8f c) {
  return __builtin_amdgcn_wmma_f32_16x16x32_bf16(
      /*neg_a=*/false, a, /*neg_b=*/false, b,
      /*c_mod=*/(short)0, c, /*reuse_a=*/false, /*reuse_b=*/false);
}

// C/D layout: VGPR r -> M = r + 8*(lane>=16), N = lane&15 (ISA 7.12.2)

// ---------------------------------------------------------------------------
// 1) LayerNorm rows of DM=1024, f32 in -> bf16 out. One 256-thread block/row.
// ---------------------------------------------------------------------------
__global__ void ln_kernel(const float* __restrict__ x, const float* __restrict__ g,
                          const float* __restrict__ bta, bf16* __restrict__ y) {
  const int row = blockIdx.x;
  const int tid = threadIdx.x;
  const float* xr = x + (size_t)row * DM;
  bf16* yr = y + (size_t)row * DM;

  float4 v = ((const float4*)xr)[tid];
  float s  = v.x + v.y + v.z + v.w;
  float ss = v.x * v.x + v.y * v.y + v.z * v.z + v.w * v.w;

  const int w = tid >> 5, lane = tid & 31;
  __shared__ float reds[8], redss[8];
#pragma unroll
  for (int off = 16; off > 0; off >>= 1) {
    s  += __shfl_xor(s, off, 32);
    ss += __shfl_xor(ss, off, 32);
  }
  if (lane == 0) { reds[w] = s; redss[w] = ss; }
  __syncthreads();
  float ts = 0.f, tss = 0.f;
#pragma unroll
  for (int i = 0; i < 8; ++i) { ts += reds[i]; tss += redss[i]; }

  const float mu  = ts * (1.0f / DM);
  const float var = tss * (1.0f / DM) - mu * mu;
  const float rs  = rsqrtf(var + 1e-5f);

  float4 gg = ((const float4*)g)[tid];
  float4 bb = ((const float4*)bta)[tid];
  const int i0 = tid * 4;
  yr[i0 + 0] = (bf16)((v.x - mu) * rs * gg.x + bb.x);
  yr[i0 + 1] = (bf16)((v.y - mu) * rs * gg.y + bb.y);
  yr[i0 + 2] = (bf16)((v.z - mu) * rs * gg.z + bb.z);
  yr[i0 + 3] = (bf16)((v.w - mu) * rs * gg.w + bb.w);
}

// ---------------------------------------------------------------------------
// 2) GEMM + bias: C[M,N] = A[M,K](bf16) @ W[K,N](f32->bf16) + bias.
//    Block tile 64x64, 8 waves (4x2), each wave two 16x16 WMMA accumulators.
// ---------------------------------------------------------------------------
template <bool F32OUT>
__global__ void gemm_bias_kernel(const bf16* __restrict__ A, const float* __restrict__ W,
                                 const float* __restrict__ bias,
                                 bf16* __restrict__ Cb, float* __restrict__ Cf,
                                 int M, int K, int N) {
  __shared__ __align__(16) bf16 As[64][40];   // [m][k]
  __shared__ __align__(16) bf16 Bs[64][40];   // [n][k]
  const int tid = threadIdx.x;
  const int gm  = blockIdx.y * 64;
  const int gn  = blockIdx.x * 64;
  const int w = tid >> 5, lane = tid & 31;
  const int wm = w & 3, wn = w >> 2;

  const int arow = tid >> 2, acol = (tid & 3) * 8;   // A: 64x32, 8 bf16/thread
  const int wk = tid >> 3, wn0 = (tid & 7) * 8;      // W: 32x64, 8 f32/thread

  v8f c0 = {}, c1 = {};
  for (int kt = 0; kt < K; kt += 32) {
    __syncthreads();
    {
      const bf16* src = A + (size_t)(gm + arow) * K + kt + acol;
#pragma unroll
      for (int i = 0; i < 8; ++i) As[arow][acol + i] = src[i];
    }
    {
      const float* src = W + (size_t)(kt + wk) * N + gn + wn0;
#pragma unroll
      for (int i = 0; i < 8; ++i) Bs[wn0 + i][wk] = (bf16)src[i];
    }
    __syncthreads();
    v16bf a  = load_frag(&As[wm * 16][0], 40);
    v16bf b0 = load_frag(&Bs[wn * 32][0], 40);
    v16bf b1 = load_frag(&Bs[wn * 32 + 16][0], 40);
    c0 = wmma_bf16(a, b0, c0);
    c1 = wmma_bf16(a, b1, c1);
  }

  const int n0 = gn + wn * 32 + (lane & 15);
  const int m0 = gm + wm * 16 + ((lane & 16) ? 8 : 0);
  const float bias0 = bias[n0], bias1 = bias[n0 + 16];
#pragma unroll
  for (int r = 0; r < 8; ++r) {
    const int m = m0 + r;
    if (F32OUT) {
      Cf[(size_t)m * N + n0]      = c0[r] + bias0;
      Cf[(size_t)m * N + n0 + 16] = c1[r] + bias1;
    } else {
      Cb[(size_t)m * N + n0]      = (bf16)(c0[r] + bias0);
      Cb[(size_t)m * N + n0 + 16] = (bf16)(c1[r] + bias1);
    }
  }
}

// ---------------------------------------------------------------------------
// 3) scores[b,h,q,k] = SCL * sum_d Q[b,q,h,d] * K[b,k,h,d].  Per-(b,h) GEMM,
//    64x64 tile, Kdim = DH = 64 (two WMMA k-steps, no k loop staging needed).
// ---------------------------------------------------------------------------
__global__ void scores_kernel(const bf16* __restrict__ Q, const bf16* __restrict__ Kmat,
                              float* __restrict__ atten) {
  __shared__ __align__(16) bf16 Qs[64][72];   // [q][d]
  __shared__ __align__(16) bf16 Ks[64][72];   // [k][d]  (B matrix as [n][kdim])
  const int tid = threadIdx.x;
  const int kt = blockIdx.x * 64;
  const int qt = blockIdx.y * 64;
  const int z  = blockIdx.z;               // b*NH + h
  const int b = z >> 4, h = z & 15;

  const int lrow = tid >> 2, lcol = (tid & 3) * 16;
  {
    const bf16* src = Q + ((size_t)(b * NQ + qt + lrow) * HD) + h * DH + lcol;
#pragma unroll
    for (int i = 0; i < 16; ++i) Qs[lrow][lcol + i] = src[i];
  }
  {
    const bf16* src = Kmat + ((size_t)(b * NKV + kt + lrow) * HD) + h * DH + lcol;
#pragma unroll
    for (int i = 0; i < 16; ++i) Ks[lrow][lcol + i] = src[i];
  }
  __syncthreads();

  const int w = tid >> 5, lane = tid & 31;
  const int wm = w & 3, wn = w >> 2;
  v8f c0 = {}, c1 = {};
#pragma unroll
  for (int d0 = 0; d0 < 64; d0 += 32) {
    v16bf a  = load_frag(&Qs[wm * 16][d0], 72);
    v16bf b0 = load_frag(&Ks[wn * 32][d0], 72);
    v16bf b1 = load_frag(&Ks[wn * 32 + 16][d0], 72);
    c0 = wmma_bf16(a, b0, c0);
    c1 = wmma_bf16(a, b1, c1);
  }

  const int n0 = kt + wn * 32 + (lane & 15);
  const int m0 = qt + wm * 16 + ((lane & 16) ? 8 : 0);
  float* dst = atten + (size_t)z * NQ * NKV;
#pragma unroll
  for (int r = 0; r < 8; ++r) {
    dst[(size_t)(m0 + r) * NKV + n0]      = c0[r] * SCL;
    dst[(size_t)(m0 + r) * NKV + n0 + 16] = c1[r] * SCL;
  }
}

// ---------------------------------------------------------------------------
// 4) rpe: atten[b,h,q,k] += SCL * sum_d Q[b,q,h,d] * rpe[q,k,d].
//    Restructured per q as GEMM: M = B*H = 32 rows (m = b*16+h), N = 64 k's.
//    rpe tile read once per block, reused by all 32 (b,h) rows via WMMA.
// ---------------------------------------------------------------------------
__global__ void rpe_kernel(const bf16* __restrict__ Q, const float* __restrict__ rpe,
                           float* __restrict__ atten) {
  __shared__ __align__(16) bf16 Qs[32][72];   // [m=b*16+h][d]
  __shared__ __align__(16) bf16 Rs[64][72];   // [k][d]
  const int tid = threadIdx.x;
  const int kt = blockIdx.x * 64;
  const int q  = blockIdx.y;
  {
    const int m = tid >> 3, d0 = (tid & 7) * 8;
    const int b = m >> 4, h = m & 15;
    const bf16* src = Q + ((size_t)(b * NQ + q) * HD) + h * DH + d0;
#pragma unroll
    for (int i = 0; i < 8; ++i) Qs[m][d0 + i] = src[i];
  }
  {
    const int kk = tid >> 2, d0 = (tid & 3) * 16;
    const float* src = rpe + ((size_t)q * NKV + kt + kk) * DH + d0;
#pragma unroll
    for (int i = 0; i < 16; ++i) Rs[kk][d0 + i] = (bf16)src[i];
  }
  __syncthreads();

  const int w = tid >> 5, lane = tid & 31;
  const int wm = w & 1, wn = w >> 1;   // 2 row-groups x 4 col-groups of 16
  v8f c = {};
#pragma unroll
  for (int d0 = 0; d0 < 64; d0 += 32) {
    v16bf a  = load_frag(&Qs[wm * 16][d0], 72);
    v16bf bf = load_frag(&Rs[wn * 16][d0], 72);
    c = wmma_bf16(a, bf, c);
  }

  const int n0 = kt + wn * 16 + (lane & 15);
  const int m0 = wm * 16 + ((lane & 16) ? 8 : 0);
#pragma unroll
  for (int r = 0; r < 8; ++r) {
    const int m = m0 + r;  // == z = b*NH + h
    const size_t idx = ((size_t)m * NQ + q) * NKV + n0;
    atten[idx] += c[r] * SCL;
  }
}

// ---------------------------------------------------------------------------
// 5) Masked softmax over k (NKV=2048), in place. One block per (b,h,q) row.
// ---------------------------------------------------------------------------
__global__ void softmax_kernel(float* __restrict__ atten,
                               const unsigned char* __restrict__ mask) {
  const int row = blockIdx.x;            // (b*NH + h)*NQ + q
  const int b = row / (NH * NQ);
  float* p = atten + (size_t)row * NKV;
  const unsigned char* mrow = mask + (size_t)b * NKV;
  const int tid = threadIdx.x;
  const int w = tid >> 5, lane = tid & 31;
  __shared__ float red[8];

  float vals[8];
  float mx = -INFINITY;
#pragma unroll
  for (int i = 0; i < 8; ++i) {
    const int k = tid + i * 256;
    const float v = mrow[k] ? p[k] : -INFINITY;
    vals[i] = v;
    mx = fmaxf(mx, v);
  }
#pragma unroll
  for (int off = 16; off > 0; off >>= 1) mx = fmaxf(mx, __shfl_xor(mx, off, 32));
  if (lane == 0) red[w] = mx;
  __syncthreads();
  float m = red[0];
#pragma unroll
  for (int i = 1; i < 8; ++i) m = fmaxf(m, red[i]);
  __syncthreads();

  float s = 0.f;
#pragma unroll
  for (int i = 0; i < 8; ++i) {
    const float e = (vals[i] == -INFINITY) ? 0.f : __expf(vals[i] - m);
    vals[i] = e;
    s += e;
  }
#pragma unroll
  for (int off = 16; off > 0; off >>= 1) s += __shfl_xor(s, off, 32);
  if (lane == 0) red[w] = s;
  __syncthreads();
  float ts = 0.f;
#pragma unroll
  for (int i = 0; i < 8; ++i) ts += red[i];
  const float inv = 1.0f / ts;
#pragma unroll
  for (int i = 0; i < 8; ++i) p[tid + i * 256] = vals[i] * inv;
}

// ---------------------------------------------------------------------------
// 6) out[b,q,h,d] = sum_k atten[b,h,q,k] * V[b,k,h,d].  Per-(b,h) GEMM,
//    M tile 64 q, N = DH = 64, k-loop over NKV in 64 steps (atten f32->bf16).
// ---------------------------------------------------------------------------
__global__ void av_kernel(const float* __restrict__ atten, const bf16* __restrict__ V,
                          bf16* __restrict__ out) {
  __shared__ __align__(16) bf16 As[64][72];   // [q][k]
  __shared__ __align__(16) bf16 Vs[64][72];   // [d][k]
  const int tid = threadIdx.x;
  const int qt = blockIdx.x * 64;
  const int z  = blockIdx.y;
  const int b = z >> 4, h = z & 15;
  const int w = tid >> 5, lane = tid & 31;
  const int wm = w & 3, wn = w >> 2;
  const int lrow = tid >> 2, lcol = (tid & 3) * 16;

  const float* abase = atten + ((size_t)z * NQ + qt) * NKV;
  const bf16*  vbase = V + (size_t)b * NKV * HD + h * DH;

  v8f c0 = {}, c1 = {};
  for (int kt = 0; kt < NKV; kt += 64) {
    __syncthreads();
    {
      const float* src = abase + (size_t)lrow * NKV + kt + lcol;
#pragma unroll
      for (int i = 0; i < 16; ++i) As[lrow][lcol + i] = (bf16)src[i];
    }
    {
      const bf16* src = vbase + (size_t)(kt + lrow) * HD + lcol;
#pragma unroll
      for (int i = 0; i < 16; ++i) Vs[lcol + i][lrow] = src[i];   // transpose k<->d
    }
    __syncthreads();
#pragma unroll
    for (int k0 = 0; k0 < 64; k0 += 32) {
      v16bf a  = load_frag(&As[wm * 16][k0], 72);
      v16bf b0 = load_frag(&Vs[wn * 32][k0], 72);
      v16bf b1 = load_frag(&Vs[wn * 32 + 16][k0], 72);
      c0 = wmma_bf16(a, b0, c0);
      c1 = wmma_bf16(a, b1, c1);
    }
  }

  const int d0 = wn * 32 + (lane & 15);
  const int m0 = qt + wm * 16 + ((lane & 16) ? 8 : 0);
#pragma unroll
  for (int r = 0; r < 8; ++r) {
    const int m = m0 + r;
    out[((size_t)(b * NQ + m)) * HD + h * DH + d0]      = (bf16)c0[r];
    out[((size_t)(b * NQ + m)) * HD + h * DH + d0 + 16] = (bf16)c1[r];
  }
}

// ---------------------------------------------------------------------------
// Launch
// ---------------------------------------------------------------------------
extern "C" void kernel_launch(void* const* d_in, const int* in_sizes, int n_in,
                              void* d_out, int out_size, void* d_ws, size_t ws_size,
                              hipStream_t stream) {
  (void)in_sizes; (void)n_in; (void)out_size; (void)ws_size;
  const float* q    = (const float*)d_in[0];
  const float* kv   = (const float*)d_in[1];
  const unsigned char* mask = (const unsigned char*)d_in[2];
  const float* rpe  = (const float*)d_in[3];
  const float* Wq   = (const float*)d_in[4];
  const float* bq   = (const float*)d_in[5];
  const float* Wk   = (const float*)d_in[6];
  const float* bk   = (const float*)d_in[7];
  const float* Wv   = (const float*)d_in[8];
  const float* bv   = (const float*)d_in[9];
  const float* Wo   = (const float*)d_in[10];
  const float* bo   = (const float*)d_in[11];
  const float* lng  = (const float*)d_in[12];
  const float* lnb  = (const float*)d_in[13];

  float* out   = (float*)d_out;
  float* atten = out + (size_t)BB * NQ * DM;   // second tuple output

  bf16* w   = (bf16*)d_ws;
  bf16* qn  = w;  w += (size_t)BB * NQ * DM;
  bf16* kvn = w;  w += (size_t)BB * NKV * DM;
  bf16* Qb  = w;  w += (size_t)BB * NQ * HD;
  bf16* Kb  = w;  w += (size_t)BB * NKV * HD;
  bf16* Vb  = w;  w += (size_t)BB * NKV * HD;
  bf16* AOb = w;  // [BB*NQ, HD]

  ln_kernel<<<BB * NQ, 256, 0, stream>>>(q, lng, lnb, qn);
  ln_kernel<<<BB * NKV, 256, 0, stream>>>(kv, lng, lnb, kvn);

  gemm_bias_kernel<false><<<dim3(HD / 64, BB * NQ / 64), 256, 0, stream>>>(
      qn, Wq, bq, Qb, nullptr, BB * NQ, DM, HD);
  gemm_bias_kernel<false><<<dim3(HD / 64, BB * NKV / 64), 256, 0, stream>>>(
      kvn, Wk, bk, Kb, nullptr, BB * NKV, DM, HD);
  gemm_bias_kernel<false><<<dim3(HD / 64, BB * NKV / 64), 256, 0, stream>>>(
      kvn, Wv, bv, Vb, nullptr, BB * NKV, DM, HD);

  scores_kernel<<<dim3(NKV / 64, NQ / 64, BB * NH), 256, 0, stream>>>(Qb, Kb, atten);
  rpe_kernel<<<dim3(NKV / 64, NQ), 256, 0, stream>>>(Qb, rpe, atten);
  softmax_kernel<<<BB * NH * NQ, 256, 0, stream>>>(atten, mask);
  av_kernel<<<dim3(NQ / 64, BB * NH), 256, 0, stream>>>(atten, Vb, AOb);

  gemm_bias_kernel<true><<<dim3(DM / 64, BB * NQ / 64), 256, 0, stream>>>(
      AOb, Wo, bo, nullptr, out, BB * NQ, HD, DM);
}
